// StyleSimilarity_21663815041349
// MI455X (gfx1250) — compile-verified, matrix operand order checked by reading, not run on hardware
//
#include <hip/hip_runtime.h>
#include <hip/hip_bf16.h>

// ---------------- problem constants ----------------
#define BB   8
#define SS   2048
#define DD   1024
#define DD2  512
#define KB   32            // keys per staged block in attention
#define ESTEPS (DD2 / 32)  // 16 K-steps of 32 over the e dimension
#define SHIFT 20.0f        // constant softmax shift (exact transform)
#define SC    0.044194173824159216f  // 1/sqrt(512)

typedef __attribute__((ext_vector_type(16))) __bf16 v16bf;
typedef __attribute__((ext_vector_type(8)))  __bf16 v8bf;
typedef __attribute__((ext_vector_type(8)))  float  v8f;

// D = A(16x32 bf16) x B(32x16 bf16) + C(16x16 f32)
__device__ inline v8f wmma_bf16(v16bf a, v16bf b, v8f c) {
    return __builtin_amdgcn_wmma_f32_16x16x32_bf16(
        /*neg_a=*/false, a, /*neg_b=*/false, b,
        /*c_mod=*/(short)0, c, /*reuse_a=*/false, /*reuse_b=*/false);
}

// Per-lane fragment gather following the documented 16-bit A/B layout:
// lanes 0-15: row = lane, K = 0..7 and 16..23 ; lanes 16-31: row = lane-16, K = 8..15 and 24..31.
// base points at [row0][k0] of a K-contiguous tile with row stride LDH halves (LDH*2 % 16 == 0).
template<int LDH>
__device__ inline v16bf ld_frag(const __bf16* base, int lr, int hb) {
    const __bf16* p = base + lr * LDH + hb * 8;
    v8bf lo = *(const v8bf*)(p);
    v8bf hi = *(const v8bf*)(p + 16);
    v16bf r;
#pragma unroll
    for (int i = 0; i < 8; ++i) { r[i] = lo[i]; r[8 + i] = hi[i]; }
    return r;
}

// low 32 bits of a flat shared-memory address == LDS byte offset (ISA 10.2)
__device__ inline unsigned lds_off(const void* p) {
    return (unsigned)(unsigned long long)p;
}

// async 16B copy: global -> LDS, tracked by ASYNCcnt (no VGPR data path)
__device__ inline void async_copy_b128(unsigned lds_addr, const void* gaddr) {
    asm volatile("global_load_async_to_lds_b128 %0, %1, off"
                 :: "v"(lds_addr), "v"(gaddr) : "memory");
}
__device__ inline void wait_asynccnt0() {
    asm volatile("s_wait_asynccnt 0x0" ::: "memory");
}

// ---------------- kernel 0: zero the w accumulators ----------------
__global__ void zero_kernel(float* p, int n) {
    int i = blockIdx.x * blockDim.x + threadIdx.x;
    if (i < n) p[i] = 0.0f;
}

// ---------------- kernel 1: projection Tp = T @ W^T (f32 -> bf16) ----------------
// M = B*S = 16384, N = 512, K = 1024. Both operands K-contiguous (NT GEMM).
// Block tile: 256(M) x 64(N), K-step 32. 8 waves, each: 2 A-frags x 4 B-frags = 8 WMMA/step.
__global__ __launch_bounds__(256)
void proj_kernel(const float* __restrict__ T1, const float* __restrict__ T2,
                 const float* __restrict__ W,
                 __bf16* __restrict__ Tp1, __bf16* __restrict__ Tp2) {
    const int which = blockIdx.z;
    const float* T = which ? T2 : T1;
    __bf16* Tp = which ? Tp2 : Tp1;
    const int bm = blockIdx.x;   // 0..63 (tiles of 256 rows)
    const int bn = blockIdx.y;   // 0..7  (tiles of 64 cols)

    const int tid  = threadIdx.x;
    const int wave = tid >> 5;
    const int lane = tid & 31;
    const int lr   = lane & 15;
    const int hb   = lane >> 4;

    __shared__ __bf16 Asb[256][40];  // 32 K + 8 pad; stride 80B (16B multiple)
    __shared__ __bf16 Bsb[64][40];

    v8f acc[2][4];
#pragma unroll
    for (int mi = 0; mi < 2; ++mi)
#pragma unroll
        for (int nf = 0; nf < 4; ++nf) acc[mi][nf] = v8f{};

    for (int kk = 0; kk < DD; kk += 32) {
        __syncthreads();
        {   // stage A: 256x32 f32 -> bf16 (thread t owns row t: 128B load, 64B store)
            const float* src = T + (size_t)(bm * 256 + tid) * DD + kk;
            __bf16* dst = &Asb[tid][0];
#pragma unroll
            for (int j = 0; j < 32; ++j) dst[j] = (__bf16)src[j];
        }
        if (tid < 64) {   // stage B: 64x32 f32 -> bf16 from W (row-major [n][k])
            const float* src = W + (size_t)(bn * 64 + tid) * DD + kk;
            __bf16* dst = &Bsb[tid][0];
#pragma unroll
            for (int j = 0; j < 32; ++j) dst[j] = (__bf16)src[j];
        }
        __syncthreads();

        // preload all fragments, then issue the WMMA chain
        v16bf af0 = ld_frag<40>(&Asb[wave * 32][0], lr, hb);
        v16bf af1 = ld_frag<40>(&Asb[wave * 32 + 16][0], lr, hb);
        v16bf bfr[4];
#pragma unroll
        for (int nf = 0; nf < 4; ++nf)
            bfr[nf] = ld_frag<40>(&Bsb[nf * 16][0], lr, hb);
#pragma unroll
        for (int nf = 0; nf < 4; ++nf) {
            acc[0][nf] = wmma_bf16(af0, bfr[nf], acc[0][nf]);
            acc[1][nf] = wmma_bf16(af1, bfr[nf], acc[1][nf]);
        }
    }

    // store D frags as bf16: row M = i + 8*hb, col N = lr
#pragma unroll
    for (int mi = 0; mi < 2; ++mi)
#pragma unroll
        for (int nf = 0; nf < 4; ++nf)
#pragma unroll
            for (int i = 0; i < 8; ++i) {
                int m = bm * 256 + wave * 32 + mi * 16 + i + 8 * hb;
                int n = bn * 64 + nf * 16 + lr;
                Tp[(size_t)m * DD2 + n] = (__bf16)acc[mi][nf][i];
            }
}

// ---------------- kernel 2: attention -> per-key weights w[b][k] ----------------
// Workgroup: batch b, 128 queries (8 waves x 16). Each wave holds its 16x512 Q strip
// as 16 resident A-frags (128 VGPRs), reused across both passes.
// Key tiles staged via ASYNC global->LDS copies (no VGPR round-trip).
// Pass 0: Z_q = sum_k exp(logit - SHIFT) (masked keys).
// Pass 1: recompute E, scale rows by mask_q/Z_q, column-sum -> atomicAdd into w[k].
__global__ __launch_bounds__(256)
void attn_kernel(const __bf16* __restrict__ Tp1, const __bf16* __restrict__ Tp2,
                 const int* __restrict__ mask1, const int* __restrict__ mask2,
                 float* __restrict__ w1, float* __restrict__ w2) {
    const int b  = blockIdx.x;
    const int qt = blockIdx.y;     // 0..15 query strips of 128
    const int which = blockIdx.z;
    const __bf16* Tp = which ? Tp2 : Tp1;
    const int* mask  = which ? mask2 : mask1;
    float* w         = which ? w2 : w1;

    const int tid  = threadIdx.x;
    const int wave = tid >> 5;
    const int lane = tid & 31;
    const int lr   = lane & 15;
    const int hb   = lane >> 4;

    __shared__ __bf16 Kt[KB][520];     // 32 keys x (512 + 8 pad), 16B-aligned rows
    __shared__ float  Zl[8][16];
    __shared__ float  invZl[8][16];

    const int q0w = qt * 128 + wave * 16;   // this wave's first query row

    // per-thread staging addresses for the key tile
    const int srow = tid >> 3;             // 0..31
    const int scol = (tid & 7) * 64;       // 0..448
    const __bf16* sbase = Tp + ((size_t)b * SS + srow) * DD2 + scol;
    unsigned sdst[8];
#pragma unroll
    for (int j = 0; j < 8; ++j) sdst[j] = lds_off(&Kt[srow][scol + j * 8]);

    // resident Q fragments: qa[t] covers e = t*32 .. t*32+31
    v16bf qa[ESTEPS];
    {
        const __bf16* qbase = Tp + ((size_t)b * SS + q0w) * DD2;
#pragma unroll
        for (int t = 0; t < ESTEPS; ++t)
            qa[t] = ld_frag<DD2>(qbase + t * 32, lr, hb);
    }

    for (int pass = 0; pass < 2; ++pass) {
        float zacc[8], invz[8];
#pragma unroll
        for (int i = 0; i < 8; ++i) { zacc[i] = 0.0f; invz[i] = 0.0f; }
        if (pass == 1) {
#pragma unroll
            for (int i = 0; i < 8; ++i) invz[i] = invZl[wave][i + 8 * hb];
        }

        for (int kb = 0; kb < SS; kb += KB) {
            __syncthreads();
            {   // async stage: 8 x 16B per thread, global -> LDS, ASYNCcnt-tracked
                const __bf16* src = sbase + (size_t)kb * DD2;
#pragma unroll
                for (int j = 0; j < 8; ++j)
                    async_copy_b128(sdst[j], src + j * 8);
                if (kb + KB < SS)   // warm L2 for the next key tile
                    __builtin_prefetch(sbase + (size_t)(kb + KB) * DD2, 0, 1);
                wait_asynccnt0();
            }
            __syncthreads();

            const int km0 = mask[b * SS + kb + lr];
            const int km1 = mask[b * SS + kb + 16 + lr];

            v8f a0 = v8f{}, a1 = v8f{};
            // software-pipelined: load B-frags for step t+1 while WMMAs of step t run
            v16bf b0 = ld_frag<520>(&Kt[0][0], lr, hb);
            v16bf b1 = ld_frag<520>(&Kt[16][0], lr, hb);
#pragma unroll
            for (int t = 0; t < ESTEPS; ++t) {
                v16bf n0 = b0, n1 = b1;
                if (t + 1 < ESTEPS) {
                    n0 = ld_frag<520>(&Kt[0][(t + 1) * 32], lr, hb);
                    n1 = ld_frag<520>(&Kt[16][(t + 1) * 32], lr, hb);
                }
                a0 = wmma_bf16(qa[t], b0, a0);
                a1 = wmma_bf16(qa[t], b1, a1);
                b0 = n0; b1 = n1;
            }

            float s0 = 0.0f, s1 = 0.0f;
#pragma unroll
            for (int i = 0; i < 8; ++i) {
                float e0 = km0 ? __expf(a0[i] * SC - SHIFT) : 0.0f;
                float e1 = km1 ? __expf(a1[i] * SC - SHIFT) : 0.0f;
                if (pass == 0) { zacc[i] += e0 + e1; }
                else           { s0 += e0 * invz[i]; s1 += e1 * invz[i]; }
            }

            if (pass == 1) {
                s0 += __shfl_xor(s0, 16, 32);
                s1 += __shfl_xor(s1, 16, 32);
                if (lane < 16) {
                    atomicAdd(&w[b * SS + kb + lane], s0);
                    atomicAdd(&w[b * SS + kb + 16 + lane], s1);
                }
            }
        }

        if (pass == 0) {
            // half-wave row sums: row q = i + 8*hb lives on 16 lanes of one half
#pragma unroll
            for (int i = 0; i < 8; ++i) {
                float r = zacc[i];
                r += __shfl_xor(r, 1, 32);
                r += __shfl_xor(r, 2, 32);
                r += __shfl_xor(r, 4, 32);
                r += __shfl_xor(r, 8, 32);
                if (lr == 0) Zl[wave][i + 8 * hb] = r;
            }
            __syncthreads();
            if (lane < 16) {
                int q = q0w + lane;
                float Z = Zl[wave][lane];
                int mq = mask[b * SS + q];
                invZl[wave][lane] = mq ? (1.0f / fmaxf(Z, 1e-30f)) : 0.0f;
            }
            __syncthreads();
        }
    }
}

// ---------------- kernel 3: v[b] = (w[b] . Tp[b]) / denom ----------------
__global__ __launch_bounds__(256)
void style_kernel(const __bf16* __restrict__ Tp1, const __bf16* __restrict__ Tp2,
                  const float* __restrict__ w1, const float* __restrict__ w2,
                  const int* __restrict__ mask1, const int* __restrict__ mask2,
                  float* __restrict__ v1, float* __restrict__ v2) {
    const int b = blockIdx.x;
    const int which = blockIdx.y;
    const __bf16* Tp = which ? Tp2 : Tp1;
    const float* w   = which ? w2 : w1;
    const int* mask  = which ? mask2 : mask1;
    float* v         = which ? v2 : v1;

    const int tid = threadIdx.x;
    __shared__ float wsh[SS];
    __shared__ float red[256];

    float mp = 0.0f;
    for (int k = tid; k < SS; k += 256) {
        wsh[k] = w[b * SS + k];
        mp += (float)mask[b * SS + k];
    }
    red[tid] = mp;
    __syncthreads();
    for (int s = 128; s > 0; s >>= 1) {
        if (tid < s) red[tid] += red[tid + s];
        __syncthreads();
    }
    const float denom = fmaxf(red[0], 1.0f);

    const __bf16* base = Tp + (size_t)b * SS * DD2;
    float a0 = 0.0f, a1 = 0.0f;
    const int e0 = tid, e1 = tid + 256;
    for (int k = 0; k < SS; ++k) {
        float wk = wsh[k];
        a0 += wk * (float)base[(size_t)k * DD2 + e0];
        a1 += wk * (float)base[(size_t)k * DD2 + e1];
    }
    v[b * DD2 + e0] = a0 / denom;
    v[b * DD2 + e1] = a1 / denom;
}

// ---------------- kernel 4: cosine similarity ----------------
__global__ __launch_bounds__(128)
void sim_kernel(const float* __restrict__ v1, const float* __restrict__ v2,
                float* __restrict__ out) {
    const int b = blockIdx.x;
    const int tid = threadIdx.x;
    __shared__ float rd[3 * 128];
    float d = 0.0f, s1 = 0.0f, s2 = 0.0f;
    for (int e = tid; e < DD2; e += 128) {
        float x = v1[b * DD2 + e], y = v2[b * DD2 + e];
        d += x * y; s1 += x * x; s2 += y * y;
    }
    rd[tid] = d; rd[128 + tid] = s1; rd[256 + tid] = s2;
    __syncthreads();
    for (int s = 64; s > 0; s >>= 1) {
        if (tid < s) {
            rd[tid]       += rd[tid + s];
            rd[128 + tid] += rd[128 + tid + s];
            rd[256 + tid] += rd[256 + tid + s];
        }
        __syncthreads();
    }
    if (tid == 0) {
        const float eps = 1e-8f;
        float n1 = fmaxf(sqrtf(rd[128]), eps);
        float n2 = fmaxf(sqrtf(rd[256]), eps);
        float sim = rd[0] / (n1 * n2);
        out[b] = (sim + 1.0f) * 0.5f;
    }
}

extern "C" void kernel_launch(void* const* d_in, const int* in_sizes, int n_in,
                              void* d_out, int out_size, void* d_ws, size_t ws_size,
                              hipStream_t stream) {
    const float* T1   = (const float*)d_in[0];
    const float* T2   = (const float*)d_in[1];
    const int*   m1   = (const int*)d_in[2];
    const int*   m2   = (const int*)d_in[3];
    const float* W    = (const float*)d_in[4];
    float* out = (float*)d_out;

    char* ws = (char*)d_ws;
    __bf16* Tp1 = (__bf16*)ws;                                   // 16 MB
    __bf16* Tp2 = (__bf16*)(ws + (size_t)16 * 1024 * 1024);      // 16 MB
    float*  w1  = (float*)(ws + (size_t)32 * 1024 * 1024);       // 64 KB
    float*  w2  = w1 + BB * SS;                                  // 64 KB
    float*  v1  = w2 + BB * SS;                                  // 16 KB
    float*  v2  = v1 + BB * DD2;                                 // 16 KB

    zero_kernel<<<dim3((2 * BB * SS + 255) / 256), dim3(256), 0, stream>>>(w1, 2 * BB * SS);
    proj_kernel<<<dim3(64, 8, 2), dim3(256), 0, stream>>>(T1, T2, W, Tp1, Tp2);
    attn_kernel<<<dim3(BB, 16, 2), dim3(256), 0, stream>>>(Tp1, Tp2, m1, m2, w1, w2);
    style_kernel<<<dim3(BB, 2), dim3(256), 0, stream>>>(Tp1, Tp2, w1, w2, m1, m2, v1, v2);
    sim_kernel<<<dim3(BB), dim3(128), 0, stream>>>(v1, v2, out);
}
